// GraphSAGE_39127152066637
// MI455X (gfx1250) — matrix operand order repaired and checked by reading, not compiled, not run on hardware
//
#include <hip/hip_runtime.h>
#include <hip/hip_bf16.h>

#define DFEAT 128

typedef __attribute__((ext_vector_type(16))) __bf16 v16bf;
typedef __attribute__((ext_vector_type(8)))  float  v8f;

// ---------- helpers ----------
__device__ __forceinline__ __bf16 f2bf(float f) {
    unsigned u = __float_as_uint(f);
    unsigned r = (u + 0x7FFFu + ((u >> 16) & 1u)) >> 16;   // round-to-nearest-even
    unsigned short s = (unsigned short)r;
    __bf16 b;
    __builtin_memcpy(&b, &s, 2);
    return b;
}

// A fragment (16x32 bf16): lane half 0 -> K = k0+{0..15}; half 1 -> K = k0+{16..31}
__device__ __forceinline__ v16bf load_a_frag(const float* __restrict__ rowp, int k0, int half) {
    const float* p = rowp + k0 + half * 16;
    v16bf f;
#pragma unroll
    for (int i = 0; i < 8; ++i) f[i] = f2bf(p[i]);
#pragma unroll
    for (int i = 0; i < 8; ++i) f[8 + i] = f2bf(p[8 + i]);
    return f;
}

// B fragment (32x16 bf16): lane n holds 16 consecutive K of column n (W row-major [out,in] -> contiguous)
__device__ __forceinline__ v16bf load_b_frag(const __bf16* __restrict__ p) {
    v16bf f;
#pragma unroll
    for (int i = 0; i < 16; ++i) f[i] = p[i];
    return f;
}

// ---------- degree over edges (structure-only, reused for all 3 layers) ----------
__global__ __launch_bounds__(256) void edge_count_kernel(const int* __restrict__ dst,
                                                         float* __restrict__ deg, int nEdges) {
    int e = blockIdx.x * blockDim.x + threadIdx.x;
    if (e < nEdges) atomicAdd(&deg[dst[e]], 1.0f);
}

// ---------- one wave per edge: gather 512B row coalesced, scatter-add ----------
__global__ __launch_bounds__(256) void edge_scatter_kernel(const float* __restrict__ feat,
                                                           const int* __restrict__ src,
                                                           const int* __restrict__ dst,
                                                           float* __restrict__ accum, int nEdges) {
    int wave = blockIdx.x * (blockDim.x >> 5) + (threadIdx.x >> 5);
    int lane = threadIdx.x & 31;
    if (wave >= nEdges) return;
    int s = src[wave];
    int d = dst[wave];
    const float4* fp = (const float4*)(feat + (size_t)s * DFEAT);
    float4 v = fp[lane];
    float* ap = accum + (size_t)d * DFEAT + lane * 4;
    atomicAdd(ap + 0, v.x);
    atomicAdd(ap + 1, v.y);
    atomicAdd(ap + 2, v.z);
    atomicAdd(ap + 3, v.w);
}

// ---------- divide segment sums by max(count,1) ----------
__global__ __launch_bounds__(256) void mean_scale_kernel(float* __restrict__ accum,
                                                         const float* __restrict__ deg, int nNodes) {
    int i = blockIdx.x * blockDim.x + threadIdx.x;
    if (i < nNodes * DFEAT) {
        float c = deg[i >> 7];
        accum[i] = accum[i] / fmaxf(c, 1.0f);
    }
}

// ---------- fused SAGE layer: Y = [relu]( AGG @ Wl^T + H @ Wr^T + b ) via bf16 WMMA ----------
__global__ __launch_bounds__(256) void sage_gemm_kernel(const float* __restrict__ H,
                                                        const float* __restrict__ AGG,
                                                        const float* __restrict__ Wl,
                                                        const float* __restrict__ Wr,
                                                        const float* __restrict__ b,
                                                        float* __restrict__ Y,
                                                        int nNodes, int doRelu) {
    __shared__ __bf16 sWl[DFEAT * DFEAT];
    __shared__ __bf16 sWr[DFEAT * DFEAT];

    int tid = threadIdx.x;
    for (int i = tid; i < DFEAT * DFEAT; i += 256) {
        sWl[i] = f2bf(Wl[i]);
        sWr[i] = f2bf(Wr[i]);
    }
    __syncthreads();

    int wave = tid >> 5;
    int lane = tid & 31;
    int half = lane >> 4;          // which half of the wave (layout split)
    int m    = lane & 15;          // A-row index within tile / B,C column index
    int rowBase = blockIdx.x * 128 + wave * 16;

    int arow = rowBase + m;        // clamp so OOB rows read valid memory (stores are guarded)
    if (arow >= nNodes) arow = nNodes - 1;
    const float* hRow = H   + (size_t)arow * DFEAT;
    const float* aRow = AGG + (size_t)arow * DFEAT;

    // 8 column tiles of 16: accumulators initialized with bias (bias depends only on N=column)
    v8f acc[8];
#pragma unroll
    for (int t = 0; t < 8; ++t) {
        float bv = b[t * 16 + m];
        v8f c;
#pragma unroll
        for (int r = 0; r < 8; ++r) c[r] = bv;
        acc[t] = c;
    }

#pragma unroll
    for (int kc = 0; kc < 4; ++kc) {
        int k0 = kc * 32;
        v16bf aAgg = load_a_frag(aRow, k0, half);
        v16bf aH   = load_a_frag(hRow, k0, half);
#pragma unroll
        for (int t = 0; t < 8; ++t) {
            int col = t * 16 + m;
            v16bf bl = load_b_frag(sWl + col * DFEAT + k0 + half * 16);
            v16bf br = load_b_frag(sWr + col * DFEAT + k0 + half * 16);
            acc[t] = __builtin_amdgcn_wmma_f32_16x16x32_bf16(
                false, aAgg, false, bl, (short)0, acc[t], false, false);
            acc[t] = __builtin_amdgcn_wmma_f32_16x16x32_bf16(
                false, aH,   false, br, (short)0, acc[t], false, false);
        }
    }

    // C/D layout: lane n (=m), VGPR r -> (M = r + 8*half, N = n)
    int mofs = half * 8;
#pragma unroll
    for (int t = 0; t < 8; ++t) {
        int col = t * 16 + m;
#pragma unroll
        for (int r = 0; r < 8; ++r) {
            int row = rowBase + mofs + r;
            if (row < nNodes) {
                float v = acc[t][r];
                if (doRelu) v = fmaxf(v, 0.0f);
                Y[(size_t)row * DFEAT + col] = v;
            }
        }
    }
}

// ---------- decoder: out[n, 0:2] = h[n,:] @ Wo^T + bo  (one wave per node) ----------
__global__ __launch_bounds__(256) void decoder_kernel(const float* __restrict__ h,
                                                      const float* __restrict__ Wo,
                                                      const float* __restrict__ bo,
                                                      float* __restrict__ out, int nNodes) {
    int wave = blockIdx.x * (blockDim.x >> 5) + (threadIdx.x >> 5);
    int lane = threadIdx.x & 31;
    if (wave >= nNodes) return;
    const float* hp = h + (size_t)wave * DFEAT;
    float p0 = 0.0f, p1 = 0.0f;
#pragma unroll
    for (int c = 0; c < 4; ++c) {
        float v = hp[lane + c * 32];
        p0 += v * Wo[lane + c * 32];
        p1 += v * Wo[DFEAT + lane + c * 32];
    }
#pragma unroll
    for (int off = 16; off > 0; off >>= 1) {
        p0 += __shfl_xor(p0, off, 32);
        p1 += __shfl_xor(p1, off, 32);
    }
    if (lane == 0) {
        out[(size_t)wave * 2 + 0] = p0 + bo[0];
        out[(size_t)wave * 2 + 1] = p1 + bo[1];
    }
}

extern "C" void kernel_launch(void* const* d_in, const int* in_sizes, int n_in,
                              void* d_out, int out_size, void* d_ws, size_t ws_size,
                              hipStream_t stream) {
    // inputs: x, edge_index, Wl1, Wr1, b1, Wl2, Wr2, b2, Wl3, Wr3, b3, Wo, bo
    const float* x    = (const float*)d_in[0];
    const int*   edge = (const int*)d_in[1];
    const float* Wl1  = (const float*)d_in[2];
    const float* Wr1  = (const float*)d_in[3];
    const float* b1   = (const float*)d_in[4];
    const float* Wl2  = (const float*)d_in[5];
    const float* Wr2  = (const float*)d_in[6];
    const float* b2   = (const float*)d_in[7];
    const float* Wl3  = (const float*)d_in[8];
    const float* Wr3  = (const float*)d_in[9];
    const float* b3   = (const float*)d_in[10];
    const float* Wo   = (const float*)d_in[11];
    const float* bo   = (const float*)d_in[12];

    const int N = in_sizes[0] / DFEAT;       // 50000
    const int E = in_sizes[1] / 2;           // 800000
    const int* srcIdx = edge;                // edge_index[0]
    const int* dstIdx = edge + E;            // edge_index[1]

    // output: [out (N,2)] ++ [h (N,128)]
    float* outp = (float*)d_out;
    float* h3   = outp + (size_t)N * 2;      // layer-3 result written directly into d_out

    // workspace: deg | agg | h1 | h2
    float* deg = (float*)d_ws;
    float* agg = deg + (((size_t)N + 255) & ~(size_t)255);
    float* h1  = agg + (size_t)N * DFEAT;
    float* h2  = h1  + (size_t)N * DFEAT;

    const size_t featBytes = (size_t)N * DFEAT * sizeof(float);
    const int scatterBlocks = (E + 7) / 8;           // 8 waves (edges) per 256-thread block
    const int scaleBlocks   = (N * DFEAT + 255) / 256;
    const int gemmBlocks    = (N + 127) / 128;

    // degree (same for all layers)
    hipMemsetAsync(deg, 0, (size_t)N * sizeof(float), stream);
    edge_count_kernel<<<(E + 255) / 256, 256, 0, stream>>>(dstIdx, deg, E);

    // layer 1: agg = mean_j x_j ; h1 = relu(agg@Wl1^T + x@Wr1^T + b1)
    hipMemsetAsync(agg, 0, featBytes, stream);
    edge_scatter_kernel<<<scatterBlocks, 256, 0, stream>>>(x, srcIdx, dstIdx, agg, E);
    mean_scale_kernel<<<scaleBlocks, 256, 0, stream>>>(agg, deg, N);
    sage_gemm_kernel<<<gemmBlocks, 256, 0, stream>>>(x, agg, Wl1, Wr1, b1, h1, N, 1);

    // layer 2
    hipMemsetAsync(agg, 0, featBytes, stream);
    edge_scatter_kernel<<<scatterBlocks, 256, 0, stream>>>(h1, srcIdx, dstIdx, agg, E);
    mean_scale_kernel<<<scaleBlocks, 256, 0, stream>>>(agg, deg, N);
    sage_gemm_kernel<<<gemmBlocks, 256, 0, stream>>>(h1, agg, Wl2, Wr2, b2, h2, N, 1);

    // layer 3 (no relu) -> writes h directly into d_out tail
    hipMemsetAsync(agg, 0, featBytes, stream);
    edge_scatter_kernel<<<scatterBlocks, 256, 0, stream>>>(h2, srcIdx, dstIdx, agg, E);
    mean_scale_kernel<<<scaleBlocks, 256, 0, stream>>>(agg, deg, N);
    sage_gemm_kernel<<<gemmBlocks, 256, 0, stream>>>(h2, agg, Wl3, Wr3, b3, h3, N, 0);

    // decoder
    decoder_kernel<<<(N + 7) / 8, 256, 0, stream>>>(h3, Wo, bo, outp, N);
}
